// rnn_12266426597892
// MI455X (gfx1250) — matrix operand order repaired and checked by reading, not compiled, not run on hardware
//
#include <hip/hip_runtime.h>
#include <cstdint>

#define SEQ   4096
#define BATCH 256
#define DIN   64
#define DH    128

// Padded LDS pitches (elements). All chosen so (row*pitch + koff)*elemsize is
// 16B aligned for b128 gathers and so the 16 lanes of a half-wave hit distinct
// LDS banks (pitch*bytes/4 mod 64 generates distinct residues over 16 rows).
#define PW_HH 136   // W_hh bf16 [128][136]
#define PW_IH 72    // W_ih bf16 [128][72]
#define PH    136   // h tile bf16 [16][136], double buffered
#define PX    68    // input tile f32 [16][68], double buffered

typedef __attribute__((ext_vector_type(16))) __bf16 bf16x16;
typedef __attribute__((ext_vector_type(8)))  __bf16 bf16x8;
typedef __attribute__((ext_vector_type(8)))  float  f32x8;
typedef __attribute__((ext_vector_type(4)))  float  f32x4;

__device__ __forceinline__ f32x8 wmma_bf16(bf16x16 a, bf16x16 b, f32x8 c) {
  // (neg_a, A, neg_b, B, c_mod, C, reuse_a, reuse_b)
  return __builtin_amdgcn_wmma_f32_16x16x32_bf16(false, a, false, b, (short)0, c,
                                                 false, false);
}

__device__ __forceinline__ float fast_tanh(float x) {
#if __has_builtin(__builtin_amdgcn_tanhf)
  return __builtin_amdgcn_tanhf(x);          // v_tanh_f32 (gfx1250 TRANS op)
#elif __has_builtin(__builtin_amdgcn_tanh_f32)
  return __builtin_amdgcn_tanh_f32(x);
#else
  return tanhf(x);
#endif
}

// LDS byte offset of a pointer that provably points into __shared__.
__device__ __forceinline__ uint32_t lds_off(const void* p) {
  return (uint32_t)(uintptr_t)(__attribute__((address_space(3))) const void*)p;
}

// GLOBAL_LOAD_ASYNC_TO_LDS_B128 (GV mode, tracked by ASYNCcnt).
__device__ __forceinline__ void async_b128_to_lds(uint32_t dst_lds, const void* g) {
  unsigned long long ga = (unsigned long long)(uintptr_t)g;
  asm volatile("global_load_async_to_lds_b128 %0, %1, off"
               :: "v"(dst_lds), "v"(ga) : "memory");
}

__device__ __forceinline__ void wait_async0() {
  asm volatile("s_wait_asynccnt 0" ::: "memory");
}

// B operand gather: 32x16 bf16 tile, elements 0..7 <-> k = kbase + 8*halfk + 0..7,
// elements 8..15 <-> k = kbase + 16 + 8*halfk + 0..7, col n fixed per lane. W stored row-major [n][k].
__device__ __forceinline__ bf16x16 load_b_tile(const __bf16* sW, int pitch,
                                               int ncol, int kbase, int halfk) {
  const __bf16* p = sW + ncol * pitch + kbase + halfk * 8;
  bf16x8 lo = *(const bf16x8*)p;
  bf16x8 hi = *(const bf16x8*)(p + 16);
  bf16x16 r;
#pragma unroll
  for (int i = 0; i < 8; ++i) { r[i] = lo[i]; r[i + 8] = hi[i]; }
  return r;
}

// A operand gather from bf16 h tile: elements 0..7 <-> k = 32kt+8*half+0..7,
// elements 8..15 <-> k = 32kt+16+8*half+0..7, row m = lane&15.
__device__ __forceinline__ bf16x16 load_a_h(const __bf16* sHt, int m, int kt, int half) {
  const __bf16* p = sHt + m * PH + kt * 32 + half * 8;
  bf16x8 lo = *(const bf16x8*)p;
  bf16x8 hi = *(const bf16x8*)(p + 16);
  bf16x16 r;
#pragma unroll
  for (int i = 0; i < 8; ++i) { r[i] = lo[i]; r[i + 8] = hi[i]; }
  return r;
}

// A operand gather from f32 input tile with f32->bf16 convert in-register.
__device__ __forceinline__ bf16x16 load_a_x(const float* sXt, int m, int kt, int half) {
  const float* p = sXt + m * PX + kt * 32 + half * 8;
  f32x4 a0 = *(const f32x4*)p;
  f32x4 a1 = *(const f32x4*)(p + 4);
  f32x4 a2 = *(const f32x4*)(p + 16);
  f32x4 a3 = *(const f32x4*)(p + 20);
  bf16x16 r;
#pragma unroll
  for (int i = 0; i < 4; ++i) {
    r[i]      = (__bf16)a0[i];
    r[i + 4]  = (__bf16)a1[i];
    r[i + 8]  = (__bf16)a2[i];
    r[i + 12] = (__bf16)a3[i];
  }
  return r;
}

// Stage input[s, rowbase:rowbase+16, 0:64] (4KB f32) into LDS via async DMA.
// 128 threads x 2 b128 each = 256 x 16B.
__device__ __forceinline__ void stage_input_async(const float* __restrict__ input,
                                                  int s, int rowbase,
                                                  float* sXbuf, int tid) {
#pragma unroll
  for (int t = 0; t < 2; ++t) {
    int idx = tid + t * 128;          // 0..255
    int r   = idx >> 4;               // row 0..15
    int kq  = (idx & 15) << 2;        // k 0..60 step 4
    const float* g = input + (((size_t)s * BATCH + rowbase + r) * DIN + kq);
    async_b128_to_lds(lds_off(&sXbuf[r * PX + kq]), g);
  }
}

__global__ __launch_bounds__(128) void rnn_fused_scan_kernel(
    const float* __restrict__ input, const float* __restrict__ Wih,
    const float* __restrict__ bih,   const float* __restrict__ Whh,
    const float* __restrict__ bhh,   float* __restrict__ out) {
  __shared__ __bf16 sWhh[DH * PW_HH];     // 34.0 KB  B for recurrent GEMM
  __shared__ __bf16 sWih[DH * PW_IH];     // 18.0 KB  B for input projection
  __shared__ __bf16 sH[2][16 * PH];       //  8.5 KB  h tile, double buffered
  __shared__ float  sX[2][16 * PX];       //  8.5 KB  input tile, double buffered

  const int tid     = threadIdx.x;
  const int wave    = tid >> 5;
  const int lane    = tid & 31;
  const int lanen   = lane & 15;          // col (B/C/D) or row (A)
  const int half    = lane >> 4;
  const int rowbase = blockIdx.x * 16;    // 16 batch rows per workgroup
  const int nt0 = wave * 2, nt1 = nt0 + 1;// this wave's two N tiles

  // One-time weight staging: f32 -> bf16, native row-major [n][k] layout.
  for (int i = tid; i < DH * DH; i += 128)
    sWhh[(i >> 7) * PW_HH + (i & 127)] = (__bf16)Whh[i];
  for (int i = tid; i < DH * DIN; i += 128)
    sWih[(i >> 6) * PW_IH + (i & 63)] = (__bf16)Wih[i];

  // Prologue: stage input for s=0,1.
  stage_input_async(input, 0, rowbase, sX[0], tid);
  stage_input_async(input, 1, rowbase, sX[1], tid);
  wait_async0();
  __syncthreads();                        // P1: weights + sX[0..1] visible

  const float bias0 = bih[nt0 * 16 + lanen] + bhh[nt0 * 16 + lanen];
  const float bias1 = bih[nt1 * 16 + lanen] + bhh[nt1 * 16 + lanen];

  // x projection for s=0 (C initialized with broadcast bias).
  f32x8 xacc0, xacc1;
#pragma unroll
  for (int v = 0; v < 8; ++v) { xacc0[v] = bias0; xacc1[v] = bias1; }
#pragma unroll
  for (int kt = 0; kt < 2; ++kt) {
    bf16x16 ax = load_a_x(sX[0], lanen, kt, half);
    xacc0 = wmma_bf16(ax, load_b_tile(sWih, PW_IH, nt0 * 16 + lanen, kt * 32, half), xacc0);
    xacc1 = wmma_bf16(ax, load_b_tile(sWih, PW_IH, nt1 * 16 + lanen, kt * 32, half), xacc1);
  }
  __syncthreads();                        // P2: separates prologue sX[0] reads
                                          // from iter-0's async write of sX[0]

  bf16x16 hA[4];                          // h_s in A layout (h_0 = 0)
#pragma unroll
  for (int kt = 0; kt < 4; ++kt)
#pragma unroll
    for (int i = 0; i < 16; ++i) hA[kt][i] = (__bf16)0.0f;

  float hout0[8], hout1[8];

  for (int s = 0; s < SEQ; ++s) {
    // ---- Phase 1: recurrent GEMM, two interleaved 4-deep WMMA chains ----
    f32x8 acc0 = xacc0, acc1 = xacc1;
#pragma unroll
    for (int kt = 0; kt < 4; ++kt) {
      bf16x16 b0 = load_b_tile(sWhh, PW_HH, nt0 * 16 + lanen, kt * 32, half);
      acc0 = wmma_bf16(hA[kt], b0, acc0);
      bf16x16 b1 = load_b_tile(sWhh, PW_HH, nt1 * 16 + lanen, kt * 32, half);
      acc1 = wmma_bf16(hA[kt], b1, acc1);
    }

    // ---- Phase 2: input projection for step s+1 (independent of h) ----
    if (s + 1 < SEQ) {
      const float* xb = sX[(s + 1) & 1];
      f32x8 x0, x1;
#pragma unroll
      for (int v = 0; v < 8; ++v) { x0[v] = bias0; x1[v] = bias1; }
#pragma unroll
      for (int kt = 0; kt < 2; ++kt) {
        bf16x16 ax = load_a_x(xb, lanen, kt, half);
        x0 = wmma_bf16(ax, load_b_tile(sWih, PW_IH, nt0 * 16 + lanen, kt * 32, half), x0);
        x1 = wmma_bf16(ax, load_b_tile(sWih, PW_IH, nt1 * 16 + lanen, kt * 32, half), x1);
      }
      xacc0 = x0; xacc1 = x1;
    }

    // ---- Phase 3: tanh, convert, store h_{s+1} to LDS (C/D layout) ----
    __bf16* hdst = sH[(s + 1) & 1];
#pragma unroll
    for (int v = 0; v < 8; ++v) {
      float h0 = fast_tanh(acc0[v]); hout0[v] = h0;
      float h1 = fast_tanh(acc1[v]); hout1[v] = h1;
      int row = v + 8 * half;
      hdst[row * PH + nt0 * 16 + lanen] = (__bf16)h0;
      hdst[row * PH + nt1 * 16 + lanen] = (__bf16)h1;
    }

    // ---- Phase 3.5: async-stage input for s+2; prefetch s+4 into L2 ----
    if (s + 2 < SEQ) {
      stage_input_async(input, s + 2, rowbase, sX[(s + 2) & 1], tid);
      if (s + 4 < SEQ) {
        const float* pf =
            input + (((size_t)(s + 4) * BATCH + rowbase + (tid >> 3)) * DIN +
                     (tid & 7) * 8);
        __builtin_prefetch(pf, 0, 0);    // global_prefetch_b8
      }
    }
    wait_async0();                       // this wave's async writes visible
    __syncthreads();                     // barrier_s: stores(h,x) -> gathers

    // ---- Phase 5: re-gather h_{s+1} into A layout for next step ----
    const __bf16* hsrc = sH[(s + 1) & 1];
#pragma unroll
    for (int kt = 0; kt < 4; ++kt) hA[kt] = load_a_h(hsrc, lanen, kt, half);
  }

  // h_final = h_SEQ (f32, pre-conversion values).
#pragma unroll
  for (int v = 0; v < 8; ++v) {
    int row = rowbase + v + 8 * half;
    out[row * DH + nt0 * 16 + lanen] = hout0[v];
    out[row * DH + nt1 * 16 + lanen] = hout1[v];
  }
}

extern "C" void kernel_launch(void* const* d_in, const int* in_sizes, int n_in,
                              void* d_out, int out_size, void* d_ws, size_t ws_size,
                              hipStream_t stream) {
  (void)in_sizes; (void)n_in; (void)out_size; (void)d_ws; (void)ws_size;
  const float* input = (const float*)d_in[0];
  const float* Wih   = (const float*)d_in[1];
  const float* bih   = (const float*)d_in[2];
  const float* Whh   = (const float*)d_in[3];
  const float* bhh   = (const float*)d_in[4];
  float* out = (float*)d_out;
  // 16 workgroups (one per 16-row batch slice) x 4 waves; persistent scan.
  rnn_fused_scan_kernel<<<BATCH / 16, 128, 0, stream>>>(input, Wih, bih, Whh, bhh, out);
}
